// TtCausalSelfAttention_67078799229283
// MI455X (gfx1250) — compile-verified, matrix operand order checked by reading, not run on hardware
//
#include <hip/hip_runtime.h>
#include <hip/hip_bf16.h>

// ---------------------------------------------------------------------------
// Causal self-attention (GPT-2 style) for MI455X / gfx1250.
// B=2, T=2048, C=768, H=12, D=64.  All matmuls on v_wmma_f32_16x16x32_f16.
// GEMM uses 32x32 register blocking per wave: 4 WMMAs per k-step on
// 2 A-frags + 2 B-frags (2x operand reuse vs 16x16 tiling).
// ---------------------------------------------------------------------------

typedef __attribute__((ext_vector_type(16))) _Float16 v16h;
typedef __attribute__((ext_vector_type(8)))  _Float16 v8h;
typedef __attribute__((ext_vector_type(8)))  float    v8f;

static constexpr int kB  = 2;
static constexpr int kT  = 2048;
static constexpr int kC  = 768;
static constexpr int kH  = 12;
static constexpr int kD  = 64;
static constexpr int k3C = 3 * kC;     // 2304
static constexpr int kM  = kB * kT;    // 4096 rows for the projections

// ---------------------------------------------------------------------------
// Fragment loader: 16-bit A-operand layout for 16x32 (MxK) tiles.
//   lanes 0-15  (half=0): row = lane,    K = {0..7, 16..23}
//   lanes 16-31 (half=1): row = lane-16, K = {8..15, 24..31}
// `p` points at (row_base + k0) of a K-contiguous row; needs 16B alignment.
// B-operand of D = A*B is fed symmetrically (as A-layout of B^T), which is
// exactly how our K-contiguous weight rows / key rows sit in memory.
// ---------------------------------------------------------------------------
__device__ __forceinline__ v16h load_frag16(const _Float16* __restrict__ p, int half) {
  const v8h* q = (const v8h*)p;
  v8h lo = q[half];       // K = 8*half + 0..7
  v8h hi = q[2 + half];   // K = 16 + 8*half + 0..7
  v16h f;
#pragma unroll
  for (int i = 0; i < 8; ++i) { f[i] = lo[i]; f[8 + i] = hi[i]; }
  return f;
}

__device__ __forceinline__ v8f wmma_f16(v16h a, v16h b, v8f c) {
  return __builtin_amdgcn_wmma_f32_16x16x32_f16(false, a, false, b, (short)0, c,
                                                false, false);
}

// ---------------------------------------------------------------------------
// fp32 -> f16 conversion
// ---------------------------------------------------------------------------
__global__ void cvt_f32_to_f16(const float* __restrict__ in,
                               _Float16* __restrict__ out, int n) {
  int i = blockIdx.x * blockDim.x + threadIdx.x;
  if (i < n) out[i] = (_Float16)in[i];
}

// ---------------------------------------------------------------------------
// GEMM:  out[m,n] = sum_k A[m,k] * W[n,k] + bias[n]
// A: [M,K] f16 row-major, W: [N,K] f16 row-major (K contiguous in both).
// Block = 256 threads = 8 waves (2 along M x 4 along N).
// Block tile 64(M) x 128(N); wave tile 32x32 (2x2 WMMA accumulators).
// Writes f32 (Cf) or f16 (Ch), whichever pointer is non-null.
// ---------------------------------------------------------------------------
__global__ __launch_bounds__(256)
void gemm_wmma_f16(const _Float16* __restrict__ A, const _Float16* __restrict__ W,
                   const float* __restrict__ bias,
                   float* __restrict__ Cf, _Float16* __restrict__ Ch,
                   int M, int N, int K) {
  const int wave = threadIdx.x >> 5;
  const int lane = threadIdx.x & 31;
  const int half = lane >> 4;
  const int r    = lane & 15;

  const int wm = wave & 1;     // 2 waves along M
  const int wn = wave >> 1;    // 4 waves along N
  const int m0 = blockIdx.y * 64 + wm * 32;
  const int n0 = blockIdx.x * 128 + wn * 32;
  if (m0 >= M || n0 >= N) return;

  const _Float16* pa0 = A + (size_t)(m0 + r) * K;        // A rows for subtile 0
  const _Float16* pa1 = A + (size_t)(m0 + 16 + r) * K;   // A rows for subtile 1
  const _Float16* pw0 = W + (size_t)(n0 + r) * K;        // W rows (B^T cols)
  const _Float16* pw1 = W + (size_t)(n0 + 16 + r) * K;

  v8f acc00 = {}, acc01 = {}, acc10 = {}, acc11 = {};
  for (int k0 = 0; k0 < K; k0 += 32) {
    __builtin_prefetch(pa0 + k0 + 128, 0, 1);            // global_prefetch_b8
    __builtin_prefetch(pw0 + k0 + 128, 0, 1);
    v16h a0 = load_frag16(pa0 + k0, half);
    v16h a1 = load_frag16(pa1 + k0, half);
    v16h b0 = load_frag16(pw0 + k0, half);
    v16h b1 = load_frag16(pw1 + k0, half);
    acc00 = wmma_f16(a0, b0, acc00);
    acc01 = wmma_f16(a0, b1, acc01);
    acc10 = wmma_f16(a1, b0, acc10);
    acc11 = wmma_f16(a1, b1, acc11);
  }

  // C layout: VGPR i -> row (subtile_m + i + 8*half), col (subtile_n + r)
  const float bv0 = bias ? bias[n0 + r] : 0.0f;
  const float bv1 = bias ? bias[n0 + 16 + r] : 0.0f;
  const v8f* accs[4] = {&acc00, &acc01, &acc10, &acc11};
#pragma unroll
  for (int t = 0; t < 4; ++t) {
    const int mi = t >> 1, ni = t & 1;
    const float bv = ni ? bv1 : bv0;
    const int nc = n0 + ni * 16 + r;
#pragma unroll
    for (int i = 0; i < 8; ++i) {
      const int m = m0 + mi * 16 + i + 8 * half;
      const float v = (*accs[t])[i] + bv;
      if (Cf) Cf[(size_t)m * N + nc] = v;
      else    Ch[(size_t)m * N + nc] = (_Float16)v;
    }
  }
}

// ---------------------------------------------------------------------------
// Flash attention over qkv[B,T,3C] (f16).  grid = (T/128, H, B), 8 waves/block,
// each wave owns a 16-row q tile and streams keys in 32-wide blocks with
// online softmax.  Per-wave LDS: P staging (16x32 f16) + V staging (32x64 f16).
// ---------------------------------------------------------------------------
__global__ __launch_bounds__(256)
void attn_wmma(const _Float16* __restrict__ qkv, _Float16* __restrict__ y) {
  __shared__ _Float16 smem[8][16 * 32 + 32 * 64];   // 40 KB

  const int wave = threadIdx.x >> 5;
  const int lane = threadIdx.x & 31;
  const int half = lane >> 4;
  const int r    = lane & 15;

  const int b  = blockIdx.z;
  const int h  = blockIdx.y;
  const int q0 = blockIdx.x * 128 + wave * 16;

  _Float16* Plds = &smem[wave][0];
  _Float16* Vlds = &smem[wave][16 * 32];

  const _Float16* qbase = qkv + (size_t)b * kT * k3C + h * kD;
  const _Float16* kbase = qbase + kC;
  const _Float16* vbase = qbase + 2 * kC;

  // Q fragments (row q0 + r, D split into two K=32 halves), loaded once.
  const _Float16* qrow = qbase + (size_t)(q0 + r) * k3C;
  const v16h qf0 = load_frag16(qrow + 0, half);
  const v16h qf1 = load_frag16(qrow + 32, half);

  v8f acc[4] = {v8f{}, v8f{}, v8f{}, v8f{}};
  float mrow[8], lrow[8];
#pragma unroll
  for (int i = 0; i < 8; ++i) { mrow[i] = -1.0e30f; lrow[i] = 0.0f; }

  const int kend = q0 + 16;           // keys needed: [0, q0+15]
  for (int kb = 0; kb < kend; kb += 32) {
    // ---- S = (Q K^T) * 1/sqrt(D) for 32 keys (two 16x16 tiles) ----
    v8f s[2];
#pragma unroll
    for (int kg = 0; kg < 2; ++kg) {
      int kc = kb + kg * 16 + r;
      kc = kc < (kT - 1) ? kc : (kT - 1);           // clamp (masked anyway)
      const _Float16* krow = kbase + (size_t)kc * k3C;
      v8f z = {};
      z = wmma_f16(qf0, load_frag16(krow + 0, half), z);
      z = wmma_f16(qf1, load_frag16(krow + 32, half), z);
      s[kg] = z;
    }

    // ---- scale + causal mask ----
#pragma unroll
    for (int kg = 0; kg < 2; ++kg)
#pragma unroll
      for (int i = 0; i < 8; ++i) {
        const int row = q0 + i + 8 * half;
        const int col = kb + kg * 16 + r;
        const float v = s[kg][i] * 0.125f;          // 1/sqrt(64)
        s[kg][i] = (col <= row) ? v : -3.0e30f;
      }

    // ---- row max across 32 cols (16 lanes per half-wave hold the cols) ----
    float tmax[8];
#pragma unroll
    for (int i = 0; i < 8; ++i) tmax[i] = fmaxf(s[0][i], s[1][i]);
    for (int off = 1; off < 16; off <<= 1)
#pragma unroll
      for (int i = 0; i < 8; ++i)
        tmax[i] = fmaxf(tmax[i], __shfl_xor(tmax[i], off, 32));

    float alpha[8];
#pragma unroll
    for (int i = 0; i < 8; ++i) {
      const float mn = fmaxf(mrow[i], tmax[i]);
      alpha[i] = __expf(mrow[i] - mn);
      mrow[i] = mn;
    }

    // ---- P = exp(S - m), partial row sums ----
    float lsum[8];
#pragma unroll
    for (int i = 0; i < 8; ++i) lsum[i] = 0.0f;
#pragma unroll
    for (int kg = 0; kg < 2; ++kg)
#pragma unroll
      for (int i = 0; i < 8; ++i) {
        const float p = __expf(s[kg][i] - mrow[i]);
        s[kg][i] = p;
        lsum[i] += p;
      }
    for (int off = 1; off < 16; off <<= 1)
#pragma unroll
      for (int i = 0; i < 8; ++i) lsum[i] += __shfl_xor(lsum[i], off, 32);
#pragma unroll
    for (int i = 0; i < 8; ++i) lrow[i] = lrow[i] * alpha[i] + lsum[i];

    // ---- rescale accumulators ----
#pragma unroll
    for (int dt = 0; dt < 4; ++dt)
#pragma unroll
      for (int i = 0; i < 8; ++i) acc[dt][i] *= alpha[i];

    // ---- stage P (C-layout -> row-major) and V tile in LDS ----
    asm volatile("s_wait_dscnt 0" ::: "memory");    // prior reads done before overwrite
#pragma unroll
    for (int kg = 0; kg < 2; ++kg)
#pragma unroll
      for (int i = 0; i < 8; ++i)
        Plds[(i + 8 * half) * 32 + kg * 16 + r] = (_Float16)s[kg][i];

    {
      int kv = kb + lane;
      kv = kv < (kT - 1) ? kv : (kT - 1);
      const _Float16* vrow = vbase + (size_t)kv * k3C;
      v8h* dst = (v8h*)&Vlds[lane * kD];
#pragma unroll
      for (int j = 0; j < 8; ++j) dst[j] = *(const v8h*)(vrow + j * 8);
    }
    asm volatile("s_wait_dscnt 0" ::: "memory");    // staging visible to gathers

    // ---- P A-fragment (row r, K=key-in-block) ----
    const v16h pf = load_frag16(Plds + r * 32, half);

    // ---- PV: 4 d-tiles, B-frag gathered (V^T columns) from LDS ----
#pragma unroll
    for (int dt = 0; dt < 4; ++dt) {
      v16h vf;
      const int d = dt * 16 + r;
#pragma unroll
      for (int j = 0; j < 8; ++j)      vf[j]     = Vlds[(8 * half + j) * kD + d];
#pragma unroll
      for (int j = 0; j < 8; ++j)      vf[8 + j] = Vlds[(16 + 8 * half + j) * kD + d];
      acc[dt] = wmma_f16(pf, vf, acc[dt]);
    }
  }

  // ---- normalize and store y[b, q, h*64 + d] as f16 ----
  float inv[8];
#pragma unroll
  for (int i = 0; i < 8; ++i) inv[i] = 1.0f / lrow[i];
  _Float16* yp = y + (size_t)b * kT * kC + h * kD;
#pragma unroll
  for (int dt = 0; dt < 4; ++dt)
#pragma unroll
    for (int i = 0; i < 8; ++i)
      yp[(size_t)(q0 + i + 8 * half) * kC + dt * 16 + r] =
          (_Float16)(acc[dt][i] * inv[i]);
}

// ---------------------------------------------------------------------------
// Host-side orchestration (stream-ordered; graph-capture safe).
// Workspace layout (f16 elements):
//   xh   [4096*768]   wah [2304*768]   wph [768*768]
//   qkvh [4096*2304]  yh  [4096*768]       total ~36.2 MB
// ---------------------------------------------------------------------------
extern "C" void kernel_launch(void* const* d_in, const int* in_sizes, int n_in,
                              void* d_out, int out_size, void* d_ws, size_t ws_size,
                              hipStream_t stream) {
  const float* x      = (const float*)d_in[0];
  const float* w_attn = (const float*)d_in[1];
  const float* b_attn = (const float*)d_in[2];
  const float* w_proj = (const float*)d_in[3];
  const float* b_proj = (const float*)d_in[4];
  float* out = (float*)d_out;

  _Float16* xh   = (_Float16*)d_ws;
  _Float16* wah  = xh + (size_t)kM * kC;
  _Float16* wph  = wah + (size_t)k3C * kC;
  _Float16* qkvh = wph + (size_t)kC * kC;
  _Float16* yh   = qkvh + (size_t)kM * k3C;

  const int nx = kM * kC, nwa = k3C * kC, nwp = kC * kC;
  cvt_f32_to_f16<<<(nx  + 255) / 256, 256, 0, stream>>>(x, xh, nx);
  cvt_f32_to_f16<<<(nwa + 255) / 256, 256, 0, stream>>>(w_attn, wah, nwa);
  cvt_f32_to_f16<<<(nwp + 255) / 256, 256, 0, stream>>>(w_proj, wph, nwp);

  // qkv = x @ w_attn^T + b_attn   (f16 epilogue)
  dim3 g1(k3C / 128, kM / 64);
  gemm_wmma_f16<<<g1, 256, 0, stream>>>(xh, wah, b_attn, nullptr, qkvh,
                                        kM, k3C, kC);

  // flash attention
  dim3 g2(kT / 128, kH, kB);
  attn_wmma<<<g2, 256, 0, stream>>>(qkvh, yh);

  // out = y @ w_proj^T + b_proj   (f32 epilogue)
  dim3 g3(kC / 128, kM / 64);
  gemm_wmma_f16<<<g3, 256, 0, stream>>>(yh, wph, b_proj, out, nullptr,
                                        kM, kC, kC);
}
